// CrossAttention_2705829396864
// MI455X (gfx1250) — matrix operand order repaired and checked by reading, not compiled, not run on hardware
//
#include <hip/hip_runtime.h>
#include <hip/hip_bf16.h>

// ---------------------------------------------------------------------------
// CDNA5 (gfx1250) cross-attention, bf16 WMMA pipeline.
//   x[4,2048,1024], context[4,2048,768], mask[4,2048] (bool),
//   W_q[1024,512], W_k[768,512], W_v[768,512], W_o[512,1024], b_o[1024]
//   out[4,2048,1024] fp32
// ---------------------------------------------------------------------------

#define BATCH 4
#define NQ    2048
#define NJ    2048
#define QDIM  1024
#define CDIM  768
#define HEADS 8
#define DHEAD 64
#define INNER 512

typedef __attribute__((ext_vector_type(16))) __bf16          v16bf;
typedef __attribute__((ext_vector_type(8)))  float           v8f;
typedef __attribute__((ext_vector_type(8)))  unsigned short  ushort8;
typedef __attribute__((ext_vector_type(4)))  float           vf4;

union BF16x16 {
    v16bf          v;
    unsigned short u[16];
    ushort8        h[2];
};

// native convert (v_cvt_pk_bf16_f32 when paired, RNE)
static __device__ __forceinline__ unsigned short f2bfu(float f) {
    return __builtin_bit_cast(unsigned short, (__bf16)f);
}

static __device__ __forceinline__ v8f wmma_bf16(v16bf a, v16bf b, v8f c) {
    // D = A(16x32 bf16) * B(32x16 bf16) + C(16x16 f32)
    return __builtin_amdgcn_wmma_f32_16x16x32_bf16(
        false, a, false, b, (short)0, c, false, false);
}

// ---------------------------------------------------------------------------
// Generic bf16 WMMA GEMM:  C[M,N] = A[M,K] * W[K,N]
//   AMODE 0: A fp32 (converted while staging)   AMODE 1: A bf16
//   OMODE 0: bf16 head-split      Obf[((b*H+h)*rpb + ni)*64 + d]
//   OMODE 1: bf16 head-split, T   Obf[((b*H+h)*64 + d)*rpb + ni]
//   OMODE 2: fp32 flat + bias     Of32[row*N + col]
//   Requires M%128==0, N%64==0, K%32==0 (true for all shapes here).
// Block: 256 threads = 8 waves; block tile 128x64; wave tile 32x32.
// ---------------------------------------------------------------------------
#define LDA 40   // ushort stride, 80 B  (16 B multiple)
#define LDB 40

template <int AMODE, int OMODE>
__global__ __launch_bounds__(256)
void gemm_bf16_wmma(const float* __restrict__ A32,
                    const unsigned short* __restrict__ A16,
                    const float* __restrict__ W,
                    unsigned short* __restrict__ Obf,
                    float* __restrict__ Of32,
                    const float* __restrict__ bias,
                    int M, int K, int N, int rows_per_batch) {
    __shared__ unsigned short As[128 * LDA];   // [row][k]
    __shared__ unsigned short Bs[64 * LDB];    // [col][k] (transposed)

    const int tid  = threadIdx.x;
    const int wave = tid >> 5;
    const int lane = tid & 31;
    const int wm   = wave & 3;
    const int wn   = wave >> 2;
    const int mlo  = lane & 15;
    const int g16  = lane >> 4;

    const int m0 = blockIdx.x * 128;
    const int n0 = blockIdx.y * 64;

    v8f acc[2][2] = {};

    for (int k0 = 0; k0 < K; k0 += 32) {
        __syncthreads();
        // --- stage A tile: 128 rows x 32 k, one 16-elem run per thread ---
        {
            const int row = tid >> 1;
            const int kb  = (tid & 1) * 16;
            const size_t g = (size_t)(m0 + row) * K + k0 + kb;   // 16-elem aligned
            unsigned short* dst = &As[row * LDA + kb];
            if (AMODE == 0) {
                const vf4* s4 = (const vf4*)(A32 + g);
                vf4 f0 = s4[0], f1 = s4[1], f2 = s4[2], f3 = s4[3];
                ushort8 h0, h1;
#pragma unroll
                for (int i = 0; i < 4; ++i) {
                    h0[i]     = f2bfu(f0[i]);
                    h0[i + 4] = f2bfu(f1[i]);
                    h1[i]     = f2bfu(f2[i]);
                    h1[i + 4] = f2bfu(f3[i]);
                }
                *(ushort8*)dst       = h0;
                *(ushort8*)(dst + 8) = h1;
            } else {
                const ushort8* s8 = (const ushort8*)(A16 + g);
                *(ushort8*)dst       = s8[0];
                *(ushort8*)(dst + 8) = s8[1];
            }
        }
        // --- stage B tile transposed: 32 k x 64 cols, 8 cols per thread ---
        {
            const int k  = tid >> 3;
            const int nb = (tid & 7) * 8;
            const vf4* src = (const vf4*)(W + (size_t)(k0 + k) * N + n0 + nb);
            vf4 b0 = src[0], b1 = src[1];
#pragma unroll
            for (int i = 0; i < 4; ++i) {
                Bs[(nb + i) * LDB + k]     = f2bfu(b0[i]);
                Bs[(nb + 4 + i) * LDB + k] = f2bfu(b1[i]);
            }
        }
        __syncthreads();

        // --- fragments (16 B aligned b128 LDS reads) + 4 WMMAs ---
        BF16x16 a[2], b[2];
#pragma unroll
        for (int im = 0; im < 2; ++im) {
            const int row = wm * 32 + im * 16 + mlo;
            a[im].h[0] = *(const ushort8*)&As[row * LDA + (g16 << 3)];
            a[im].h[1] = *(const ushort8*)&As[row * LDA + 16 + (g16 << 3)];
        }
#pragma unroll
        for (int in = 0; in < 2; ++in) {
            const int col = wn * 32 + in * 16 + mlo;
            const ushort8* base = (const ushort8*)&Bs[col * LDB + (g16 << 3)];
            b[in].h[0] = base[0];
            b[in].h[1] = base[2];
        }
#pragma unroll
        for (int im = 0; im < 2; ++im)
#pragma unroll
            for (int in = 0; in < 2; ++in)
                acc[im][in] = wmma_bf16(a[im].v, b[in].v, acc[im][in]);
    }

    // --- epilogue (branch-free per instantiation) ---
    const int heads = N >> 6;
#pragma unroll
    for (int im = 0; im < 2; ++im) {
#pragma unroll
        for (int in = 0; in < 2; ++in) {
#pragma unroll
            for (int r = 0; r < 8; ++r) {
                const int row = m0 + wm * 32 + im * 16 + (g16 << 3) + r;
                const int col = n0 + wn * 32 + in * 16 + mlo;
                const float val = acc[im][in][r];
                if (OMODE == 2) {
                    Of32[(size_t)row * N + col] = val + bias[col];
                } else {
                    const int bidx = row / rows_per_batch;
                    const int ni   = row % rows_per_batch;
                    const int h    = col >> 6;
                    const int d    = col & 63;
                    const size_t o = (OMODE == 0)
                        ? (((size_t)bidx * heads + h) * rows_per_batch + ni) * 64 + d
                        : (((size_t)bidx * heads + h) * 64 + d) * rows_per_batch + ni;
                    Obf[o] = f2bfu(val);
                }
            }
        }
    }
}

// ---------------------------------------------------------------------------
// Flash attention core (bf16 WMMA, fp32 online softmax).
//   Q : bf16 [b][h][row][64]      K : bf16 [b][h][j][64]
//   Vt: bf16 [b][h][64][j]  (transposed -> contiguous B fragments)
//   mask : bool [b][j]            Oa : bf16 [b][row][512]
// Grid: (NQ/128, BATCH*HEADS), block 256 = 8 waves, wave = 16 query rows.
// K/V tiles staged with GLOBAL_LOAD_ASYNC_TO_LDS_B128 (ASYNCcnt).
// ---------------------------------------------------------------------------
#define KSTR 72   // 144 B rows (16 B multiple)
#define VSTR 40   // 80 B rows
#define PSTR 40

__global__ __launch_bounds__(256)
void flash_attn_wmma(const unsigned short* __restrict__ Q,
                     const unsigned short* __restrict__ Kc,
                     const unsigned short* __restrict__ Vt,
                     const unsigned char* __restrict__ mask,
                     unsigned short* __restrict__ Oa) {
    __shared__ unsigned short Ks[32 * KSTR];       // [jrow][d]
    __shared__ unsigned short Vs[64 * VSTR];       // [d][j]
    __shared__ unsigned short Ps[8 * 16 * PSTR];   // per-wave P tile

    const int tid  = threadIdx.x;
    const int wave = tid >> 5;
    const int lane = tid & 31;
    const int mlo  = lane & 15;
    const int g16  = lane >> 4;

    const int bh = blockIdx.y;
    const int bi = bh >> 3;
    const int h  = bh & 7;
    const size_t kbase  = (size_t)bh * NJ * DHEAD;  // K  [j][d]
    const size_t vtbase = (size_t)bh * DHEAD * NJ;  // Vt [d][j]
    const int qrow0 = blockIdx.x * 128 + wave * 16;

    const float scale = 0.125f;                     // 64^-0.5

    // --- preload Q fragments: 16 rows x 64 d, two K=32 chunks ---
    BF16x16 aQ[2];
    {
        const unsigned short* qp =
            Q + (size_t)bh * NQ * DHEAD + (size_t)(qrow0 + mlo) * DHEAD;
#pragma unroll
        for (int c = 0; c < 2; ++c) {
            aQ[c].h[0] = *(const ushort8*)(qp + c * 32 + (g16 << 3));
            aQ[c].h[1] = *(const ushort8*)(qp + c * 32 + 16 + (g16 << 3));
        }
    }

    float mrow[8], lrow[8];
    v8f O[4] = {};
#pragma unroll
    for (int r = 0; r < 8; ++r) { mrow[r] = -3.0e38f; lrow[r] = 0.f; }

    for (int j0 = 0; j0 < NJ; j0 += 32) {
        __syncthreads();   // everyone done reading previous tiles
        // --- async staging: one b128 chunk per thread per tile ---
        {
            // K tile: rows j0..j0+31 of [j][64] -> contiguous 4 KB region
            const unsigned short* gk = Kc + kbase + (size_t)j0 * DHEAD + (size_t)tid * 8;
            unsigned lk = (unsigned)(size_t)&Ks[(tid >> 3) * KSTR + (tid & 7) * 8];
            asm volatile("global_load_async_to_lds_b128 %0, %1, off"
                         :: "v"(lk), "v"((unsigned long long)(size_t)gk) : "memory");
            // Vt tile: 64 rows (d) x 32 j, 64 B per row -> 4 chunks/row
            const unsigned short* gv =
                Vt + vtbase + (size_t)(tid >> 2) * NJ + j0 + (tid & 3) * 8;
            unsigned lv = (unsigned)(size_t)&Vs[(tid >> 2) * VSTR + (tid & 3) * 8];
            asm volatile("global_load_async_to_lds_b128 %0, %1, off"
                         :: "v"(lv), "v"((unsigned long long)(size_t)gv) : "memory");
            asm volatile("s_wait_asynccnt 0x0" ::: "memory");
        }
        __syncthreads();

        // --- S = Q * K^T : two 16x16 tiles, 4 WMMAs ---
        v8f s[2] = {};
#pragma unroll
        for (int c = 0; c < 2; ++c) {
#pragma unroll
            for (int jt = 0; jt < 2; ++jt) {
                BF16x16 bk;
                const int col = jt * 16 + mlo;
                const ushort8* base =
                    (const ushort8*)&Ks[col * KSTR + c * 32 + (g16 << 3)];
                bk.h[0] = base[0];
                bk.h[1] = base[2];
                s[jt] = wmma_bf16(aQ[c].v, bk.v, s[jt]);
            }
        }

        // --- scale + mask ---
#pragma unroll
        for (int jt = 0; jt < 2; ++jt) {
            const unsigned char mv = mask[(size_t)bi * NJ + j0 + jt * 16 + mlo];
#pragma unroll
            for (int r = 0; r < 8; ++r)
                s[jt][r] = mv ? (s[jt][r] * scale) : -3.0e38f;
        }

        // --- online softmax per row (16-lane reductions) ---
        float corr[8];
#pragma unroll
        for (int r = 0; r < 8; ++r) {
            float mx = fmaxf(s[0][r], s[1][r]);
#pragma unroll
            for (int off = 8; off >= 1; off >>= 1)
                mx = fmaxf(mx, __shfl_xor(mx, off, 32));
            const float mnew = fmaxf(mrow[r], mx);
            const float p0 = (s[0][r] < -1.0e37f) ? 0.f : __expf(s[0][r] - mnew);
            const float p1 = (s[1][r] < -1.0e37f) ? 0.f : __expf(s[1][r] - mnew);
            float ps = p0 + p1;
#pragma unroll
            for (int off = 8; off >= 1; off >>= 1)
                ps += __shfl_xor(ps, off, 32);
            corr[r] = __expf(mrow[r] - mnew);
            lrow[r] = lrow[r] * corr[r] + ps;
            mrow[r] = mnew;
            s[0][r] = p0;
            s[1][r] = p1;
        }
#pragma unroll
        for (int dt = 0; dt < 4; ++dt)
#pragma unroll
            for (int r = 0; r < 8; ++r) O[dt][r] *= corr[r];

        // --- P: C-layout -> A-fragment layout via wave-private LDS ---
        unsigned short* P = &Ps[wave * 16 * PSTR];
#pragma unroll
        for (int jt = 0; jt < 2; ++jt)
#pragma unroll
            for (int r = 0; r < 8; ++r)
                P[((g16 << 3) + r) * PSTR + jt * 16 + mlo] = f2bfu(s[jt][r]);
        // same-wave LDS ops are in-order (DScnt); compiler inserts waits.
        BF16x16 ap;
        ap.h[0] = *(const ushort8*)&P[mlo * PSTR + (g16 << 3)];
        ap.h[1] = *(const ushort8*)&P[mlo * PSTR + 16 + (g16 << 3)];

        // --- O += P * V : 4 WMMAs over d tiles (contiguous Vt fragments) ---
#pragma unroll
        for (int dt = 0; dt < 4; ++dt) {
            BF16x16 bv;
            const int col = dt * 16 + mlo;                 // d within head
            const ushort8* base = (const ushort8*)&Vs[col * VSTR + (g16 << 3)];
            bv.h[0] = base[0];
            bv.h[1] = base[2];
            O[dt] = wmma_bf16(ap.v, bv.v, O[dt]);
        }
    }

    // --- normalize and write out: [b][row][h*64 + d], bf16 ---
#pragma unroll
    for (int r = 0; r < 8; ++r) {
        const float inv = (lrow[r] > 0.f) ? (1.f / lrow[r]) : 0.f;
        const int row = qrow0 + (g16 << 3) + r;
#pragma unroll
        for (int dt = 0; dt < 4; ++dt) {
            const int col = h * 64 + dt * 16 + mlo;
            Oa[((size_t)bi * NQ + row) * INNER + col] = f2bfu(O[dt][r] * inv);
        }
    }
}

// ---------------------------------------------------------------------------
extern "C" void kernel_launch(void* const* d_in, const int* in_sizes, int n_in,
                              void* d_out, int out_size, void* d_ws, size_t ws_size,
                              hipStream_t stream) {
    (void)in_sizes; (void)n_in; (void)out_size; (void)ws_size;

    const float*         x    = (const float*)d_in[0];          // [4,2048,1024]
    const float*         ctx  = (const float*)d_in[1];          // [4,2048,768]
    const unsigned char* mask = (const unsigned char*)d_in[2];  // [4,2048] bool
    const float*         Wq   = (const float*)d_in[3];          // [1024,512]
    const float*         Wk   = (const float*)d_in[4];          // [768,512]
    const float*         Wv   = (const float*)d_in[5];          // [768,512]
    const float*         Wo   = (const float*)d_in[6];          // [512,1024]
    const float*         bo   = (const float*)d_in[7];          // [1024]
    float*               out  = (float*)d_out;                  // [4,2048,1024]

    const size_t per = (size_t)BATCH * HEADS * NQ * DHEAD;      // 4.19M elems
    unsigned short* q_ws = (unsigned short*)d_ws;               // [b][h][n][64]
    unsigned short* k_ws = q_ws + per;                          // [b][h][j][64]
    unsigned short* v_ws = k_ws + per;                          // [b][h][64][j]
    unsigned short* a_ws = v_ws + per;                          // [b][n][512]

    const int M = BATCH * NQ;                                   // 8192
    dim3 blk(256);

    // Q = x @ W_q  -> head-split bf16
    gemm_bf16_wmma<0, 0><<<dim3(M / 128, INNER / 64), blk, 0, stream>>>(
        x, nullptr, Wq, q_ws, nullptr, nullptr, M, QDIM, INNER, NQ);
    // K = ctx @ W_k
    gemm_bf16_wmma<0, 0><<<dim3(M / 128, INNER / 64), blk, 0, stream>>>(
        ctx, nullptr, Wk, k_ws, nullptr, nullptr, M, CDIM, INNER, NJ);
    // V = ctx @ W_v  -> transposed head-split bf16
    gemm_bf16_wmma<0, 1><<<dim3(M / 128, INNER / 64), blk, 0, stream>>>(
        ctx, nullptr, Wv, v_ws, nullptr, nullptr, M, CDIM, INNER, NJ);

    // attention core
    flash_attn_wmma<<<dim3(NQ / 128, BATCH * HEADS), blk, 0, stream>>>(
        q_ws, k_ws, v_ws, mask, a_ws);

    // out = attn @ W_o + b_o  (fp32 output)
    gemm_bf16_wmma<1, 2><<<dim3(M / 128, QDIM / 64), blk, 0, stream>>>(
        nullptr, a_ws, Wo, nullptr, out, bo, M, INNER, QDIM, NQ);
}
